// VariantDecoder_65652870086782
// MI455X (gfx1250) — matrix operand order repaired and checked by reading, not compiled, not run on hardware
//
#include <hip/hip_runtime.h>
#include <hip/hip_bf16.h>
#include <math.h>

// ---------------------------------------------------------------------------
// MI455X (gfx1250) implementation.
//
//   controller : s = sigmoid(relu(x@Ws1+bs1)@Ws2+bs2) -> k[row] (int)
//   gemm_bf16x3: h1 = relu(x @W1 +b1)   M=256 K=1024 N=2048
//   gemm_bf16x3: h2 = relu(h1@W2 +b2)   M=256 K=2048 N=1024
//   gemm_bf16x3: z  =      h2@Wd +bd    M=256 K=1024 N=50000  (sigmoid is
//                                        monotone => same ranks as probs)
//   topk_mask  : per-row radix-select threshold + stable tie-break mask
//
// GEMM: fp32 operands split on the fly into bf16 hi/lo (bf16x3:
// Ah*Bh + Ah*Bl + Al*Bh accumulated in f32 WMMA) => ~fp32 accuracy on the
// CDNA5 matrix pipe. One block owns the full M=256 strip of a BN=64 column
// block, so Wd (204.8 MB, the 23.3 TB/s HBM-bound term) streams from HBM
// exactly once. LDS reads are b128, A-tile stores b64, B-tile stores b16; fragments are register-resident
// (B fragments loaded per-N-tile to keep live VGPRs ~120, no spills).
// ---------------------------------------------------------------------------

typedef __attribute__((ext_vector_type(16))) __bf16   v16bf;
typedef __attribute__((ext_vector_type(8)))  float    v8f;
typedef __attribute__((ext_vector_type(4)))  unsigned u32x4;
typedef __attribute__((ext_vector_type(2)))  unsigned u32x2;

#define BM 256
#define BN 64
#define BK 32
#define BKP 40   // LDS K-stride in elements: 80 B rows -> all chunks 16B-aligned

union Frag {
  u32x4 q[2];
  v16bf v;
};

__device__ __forceinline__ unsigned short bits_of(__bf16 h) {
  union { __bf16 h; unsigned short s; } u;
  u.h = h;
  return u.s;
}

__device__ __forceinline__ void split_bf16(float v, __bf16& hi, __bf16& lo) {
  __bf16 h = (__bf16)v;           // RNE convert
  hi = h;
  lo = (__bf16)(v - (float)h);    // exact residual, then RNE
}

// C = act(A[256,K] @ B[K,N] + bias). K % 32 == 0, N % 4 == 0.
__global__ __launch_bounds__(256) void gemm_bf16x3(
    const float* __restrict__ A, const float* __restrict__ B,
    const float* __restrict__ bias, float* __restrict__ C,
    int K, int N, int do_relu) {
  __shared__ __bf16 Ah[BM * BKP];
  __shared__ __bf16 Al[BM * BKP];
  __shared__ __bf16 Bh[BN * BKP];
  __shared__ __bf16 Bl[BN * BKP];

  const int tid  = threadIdx.x;
  const int lane = tid & 31;
  const int wave = tid >> 5;        // wave w owns rows [32w, 32w+32)
  const int n0   = blockIdx.x * BN;

  const int half = lane >> 4;       // 0: lanes 0-15, 1: lanes 16-31
  const int lrow = lane & 15;
  const bool full_blk = (n0 + BN) <= N;   // block-uniform tail test

  v8f acc[2][4];
#pragma unroll
  for (int i = 0; i < 2; ++i)
#pragma unroll
    for (int j = 0; j < 4; ++j)
#pragma unroll
      for (int r = 0; r < 8; ++r) acc[i][j][r] = 0.0f;

  const int ksteps = K / BK;
  for (int kb = 0; kb < ksteps; ++kb) {
    const int kof = kb * BK;

    // ---- A tile: 256x32 fp32 = 2048 float4, 8/thread; hi/lo split; b64 LDS stores
#pragma unroll
    for (int i = 0; i < 8; ++i) {
      int e  = tid + i * 256;                    // 0..2047
      int m  = e >> 3;
      int kq = (e & 7) << 2;
      const float4 v = *(const float4*)(A + (size_t)m * K + kof + kq);
      float vv[4] = {v.x, v.y, v.z, v.w};
      __bf16 h[4], l[4];
#pragma unroll
      for (int j = 0; j < 4; ++j) split_bf16(vv[j], h[j], l[j]);
      u32x2 ph, pl;
      ph[0] = ((unsigned)bits_of(h[1]) << 16) | bits_of(h[0]);
      ph[1] = ((unsigned)bits_of(h[3]) << 16) | bits_of(h[2]);
      pl[0] = ((unsigned)bits_of(l[1]) << 16) | bits_of(l[0]);
      pl[1] = ((unsigned)bits_of(l[3]) << 16) | bits_of(l[2]);
      *(u32x2*)&Ah[m * BKP + kq] = ph;           // 8B-aligned (80*m + 2*kq)
      *(u32x2*)&Al[m * BKP + kq] = pl;
    }
    // ---- B tile: 32x64 fp32 = 512 float4, 2/thread; stored transposed [n][k]
#pragma unroll
    for (int i = 0; i < 2; ++i) {
      int e  = tid + i * 256;                    // 0..511
      int kk = e >> 4;
      int nq = (e & 15) << 2;
      int n  = n0 + nq;
      float vv[4];
      if (full_blk) {
        const float4 v = *(const float4*)(B + (size_t)(kof + kk) * N + n);
        vv[0] = v.x; vv[1] = v.y; vv[2] = v.z; vv[3] = v.w;
      } else {
#pragma unroll
        for (int j = 0; j < 4; ++j)
          vv[j] = (n + j < N) ? B[(size_t)(kof + kk) * N + n + j] : 0.0f;
      }
#pragma unroll
      for (int j = 0; j < 4; ++j) {
        __bf16 h, l; split_bf16(vv[j], h, l);
        Bh[(nq + j) * BKP + kk] = h;             // transposed scatter (small tile)
        Bl[(nq + j) * BKP + kk] = l;
      }
    }
    __syncthreads();

    // ---- A fragments (contiguous k order, same as B):
    // lanes 0-15: K[0..15]; lanes 16-31: K[16..31]
    Frag afh[2], afl[2];
#pragma unroll
    for (int i = 0; i < 2; ++i) {
      int m  = (2 * wave + i) * 16 + lrow;
      int c0 = half * 16;
      int c1 = half * 16 + 8;
      afh[i].q[0] = *(const u32x4*)&Ah[m * BKP + c0];   // all 16B-aligned
      afh[i].q[1] = *(const u32x4*)&Ah[m * BKP + c1];
      afl[i].q[0] = *(const u32x4*)&Al[m * BKP + c0];
      afl[i].q[1] = *(const u32x4*)&Al[m * BKP + c1];
    }

    // ---- per-N-tile B fragments + 6 WMMA each (keeps live VGPRs low)
    // B layout: lane = column; lanes 0-15 K[0..15], lanes 16-31 K[16..31]
#pragma unroll
    for (int jt = 0; jt < 4; ++jt) {
      int n = jt * 16 + lrow;
      int c = half * 16;
      Frag bh, bl;
      bh.q[0] = *(const u32x4*)&Bh[n * BKP + c];
      bh.q[1] = *(const u32x4*)&Bh[n * BKP + c + 8];
      bl.q[0] = *(const u32x4*)&Bl[n * BKP + c];
      bl.q[1] = *(const u32x4*)&Bl[n * BKP + c + 8];
#pragma unroll
      for (int i = 0; i < 2; ++i) {
        acc[i][jt] = __builtin_amdgcn_wmma_f32_16x16x32_bf16(
            false, afh[i].v, false, bh.v, (short)0, acc[i][jt], false, false);
        acc[i][jt] = __builtin_amdgcn_wmma_f32_16x16x32_bf16(
            false, afh[i].v, false, bl.v, (short)0, acc[i][jt], false, false);
        acc[i][jt] = __builtin_amdgcn_wmma_f32_16x16x32_bf16(
            false, afl[i].v, false, bh.v, (short)0, acc[i][jt], false, false);
      }
    }
    __syncthreads();
  }

  // ---- epilogue: bias + optional relu; C layout: VGPR r -> M = r + 8*half
#pragma unroll
  for (int i = 0; i < 2; ++i) {
    int mt = 2 * wave + i;
#pragma unroll
    for (int jt = 0; jt < 4; ++jt) {
      int n = n0 + jt * 16 + lrow;
      if (n < N) {
        float bv = bias[n];
#pragma unroll
        for (int r = 0; r < 8; ++r) {
          int m = mt * 16 + r + half * 8;
          float v = acc[i][jt][r] + bv;
          if (do_relu) v = fmaxf(v, 0.0f);
          C[(size_t)m * N + n] = v;
        }
      }
    }
  }
}

// ---------------------------------------------------------------------------
// Sparsity controller: one block per row, 128 threads (one per hidden unit).
// ---------------------------------------------------------------------------
__global__ __launch_bounds__(128) void controller_kernel(
    const float* __restrict__ x, const float* __restrict__ Ws1,
    const float* __restrict__ bs1, const float* __restrict__ Ws2,
    const float* __restrict__ bs2, int* __restrict__ krow) {
  const int row = blockIdx.x;
  const int tid = threadIdx.x;
  const float* xr = x + (size_t)row * 1024;

  float acc = 0.0f;
  for (int kk = 0; kk < 1024; ++kk)
    acc = fmaf(xr[kk], Ws1[(size_t)kk * 128 + tid], acc);
  float h = fmaxf(acc + bs1[tid], 0.0f);

  __shared__ float red[128];
  red[tid] = h * Ws2[tid];
  __syncthreads();
  for (int off = 64; off > 0; off >>= 1) {
    if (tid < off) red[tid] += red[tid + off];
    __syncthreads();
  }
  if (tid == 0) {
    float s    = 1.0f / (1.0f + expf(-(red[0] + bs2[0])));
    float rate = 0.005f + s * 0.095f;
    int   k    = (int)floorf(50000.0f * rate);
    krow[row]  = (k < 1) ? 1 : k;
  }
}

// ---------------------------------------------------------------------------
// Per-row top-k mask. One block (256 threads) per row. 4x8-bit radix select
// on order-preserving u32 keys; ties at threshold taken lowest-index-first
// (matches JAX stable double-argsort ranks).
// ---------------------------------------------------------------------------
#define NCOLS 50000

__device__ __forceinline__ unsigned f2ord(float f) {
  unsigned u = __float_as_uint(f);
  return u ^ ((unsigned)((int)u >> 31) | 0x80000000u);   // ascending order map
}

__global__ __launch_bounds__(256) void topk_mask_kernel(
    const float* __restrict__ z, const int* __restrict__ krow,
    float* __restrict__ out) {
  const int row = blockIdx.x;
  const int tid = threadIdx.x;
  const float* zr = z + (size_t)row * NCOLS;
  float* orow     = out + (size_t)row * NCOLS;

  __shared__ int hist[256];
  __shared__ int s_bin;
  __shared__ int s_gt;
  __shared__ int eqc[256];

  int      kk     = krow[row];
  unsigned prefix = 0;

  for (int p = 0; p < 4; ++p) {
    const int shift = 24 - 8 * p;
    hist[tid] = 0;
    __syncthreads();
    for (int j = tid; j < NCOLS; j += 256) {
      unsigned u = f2ord(zr[j]);
      if (p == 0 || (u >> (32 - 8 * p)) == prefix)
        atomicAdd(&hist[(u >> shift) & 255], 1);
    }
    __syncthreads();
    if (tid == 0) {
      int acc = 0, b = 255;
      for (; b > 0; --b) {
        if (acc + hist[b] >= kk) break;
        acc += hist[b];
      }
      s_bin = b;
      s_gt  = acc;
    }
    __syncthreads();
    prefix = (prefix << 8) | (unsigned)s_bin;
    kk -= s_gt;                       // still-needed count within chosen bin
    __syncthreads();
  }
  const unsigned uthr = prefix;       // k-th largest key; kk = #ties to take

  // Stable tie-break: chunk-per-thread prefix scan of equal-count.
  const int CH    = (NCOLS + 255) / 256;   // 196
  const int begin = tid * CH;
  const int end   = min(NCOLS, begin + CH);
  int eq = 0;
  for (int j = begin; j < end; ++j)
    if (f2ord(zr[j]) == uthr) eq++;
  eqc[tid] = eq;
  __syncthreads();
  if (tid == 0) {
    int run = 0;
    for (int t = 0; t < 256; ++t) { int v = eqc[t]; eqc[t] = run; run += v; }
  }
  __syncthreads();
  int base = eqc[tid], taken = 0;
  for (int j = begin; j < end; ++j) {
    unsigned u = f2ord(zr[j]);
    float m;
    if (u > uthr)        m = 1.0f;
    else if (u == uthr) { m = (base + taken < kk) ? 1.0f : 0.0f; taken++; }
    else                 m = 0.0f;
    orow[j] = m;
  }
}

// ---------------------------------------------------------------------------
extern "C" void kernel_launch(void* const* d_in, const int* in_sizes, int n_in,
                              void* d_out, int out_size, void* d_ws, size_t ws_size,
                              hipStream_t stream) {
  (void)in_sizes; (void)n_in; (void)out_size; (void)ws_size;
  const float* x   = (const float*)d_in[0];
  const float* W1  = (const float*)d_in[1];
  const float* b1  = (const float*)d_in[2];
  const float* W2  = (const float*)d_in[3];
  const float* b2  = (const float*)d_in[4];
  const float* Wd  = (const float*)d_in[5];
  const float* bd  = (const float*)d_in[6];
  const float* Ws1 = (const float*)d_in[7];
  const float* bs1 = (const float*)d_in[8];
  const float* Ws2 = (const float*)d_in[9];
  const float* bs2 = (const float*)d_in[10];

  char*  ws   = (char*)d_ws;
  float* z    = (float*)ws;                                   // 256*50000*4
  float* h1   = (float*)(ws + (size_t)256 * NCOLS * 4);       // 256*2048*4
  float* h2   = h1 + (size_t)256 * 2048;                      // 256*1024*4
  int*   krow = (int*)(h2 + (size_t)256 * 1024);              // 256*4

  controller_kernel<<<dim3(256), dim3(128), 0, stream>>>(x, Ws1, bs1, Ws2, bs2, krow);
  gemm_bf16x3<<<dim3(2048 / BN), dim3(256), 0, stream>>>(x,  W1, b1, h1, 1024, 2048, 1);
  gemm_bf16x3<<<dim3(1024 / BN), dim3(256), 0, stream>>>(h1, W2, b2, h2, 2048, 1024, 1);
  gemm_bf16x3<<<dim3((NCOLS + BN - 1) / BN), dim3(256), 0, stream>>>(h2, Wd, bd, z, 1024, NCOLS, 0);
  topk_mask_kernel<<<dim3(256), dim3(256), 0, stream>>>(z, krow, (float*)d_out);
}
